// NeighborAttention_17617955848670
// MI455X (gfx1250) — matrix operand order repaired and checked by reading, not compiled, not run on hardware
//
#include <hip/hip_runtime.h>

// ---------------------------------------------------------------------------
// NeighborAttention for MI455X (gfx1250, wave32).
// B=4, N=10000, K=32 neighbors, NUM_HIDDEN=128, NUM_IN=384, 4 heads x d=32.
// HBM-bound (~2 GB h_E stream @ 23.3 TB/s ~ 86us floor); all GEMM FLOPs on
// v_wmma_f32_16x16x32_bf16 (f32 accumulate).
// ---------------------------------------------------------------------------

#define KNB     32      // neighbors per node
#define NH      128     // hidden
#define NIN     384     // edge feature dim
#define NHEADS  4
#define DH      32      // head dim

#define HE_STRIDE 392   // 384 + 8 bf16 pad -> conflict-free A-fragment reads
#define KV_STRIDE 132   // 128 + 4 f32 pad

typedef __attribute__((ext_vector_type(16))) __bf16 v16bf;
typedef __attribute__((ext_vector_type(8)))  __bf16 v8bf;
typedef __attribute__((ext_vector_type(8)))  float  v8f;

// ---- prep: W_K / W_V -> bf16 (row-major, K contiguous = WMMA B fragments),
//            W_Q / W_O -> f32 transpose (coalesced broadcast reads) ----------
__global__ void prep_weights(const float* __restrict__ Wq, const float* __restrict__ Wk,
                             const float* __restrict__ Wv, const float* __restrict__ Wo,
                             __bf16* __restrict__ wkb, __bf16* __restrict__ wvb,
                             float* __restrict__ wqT, float* __restrict__ woT) {
    int id = blockIdx.x * blockDim.x + threadIdx.x;
    if (id < NH * NIN) {
        wkb[id] = (__bf16)Wk[id];
        wvb[id] = (__bf16)Wv[id];
    }
    if (id < NH * NH) {
        int j = id >> 7;        // inner index
        int c = id & 127;       // output channel
        wqT[id] = Wq[c * NH + j];   // wqT[j][c] = W_Q[c][j]
        woT[id] = Wo[c * NH + j];
    }
}

// ---- main: one workgroup (8 waves) per node --------------------------------
__global__ __launch_bounds__(256) void neigh_attn(
    const float* __restrict__ hV, const float* __restrict__ hE,
    const int* __restrict__ mask,
    const __bf16* __restrict__ wkb, const __bf16* __restrict__ wvb,
    const float* __restrict__ wqT, const float* __restrict__ woT,
    float* __restrict__ out)
{
    __shared__ __align__(16) __bf16 sh_hE[KNB * HE_STRIDE];   // 25088 B
    __shared__ float sh_K[KNB * KV_STRIDE];                   // 16896 B
    __shared__ float sh_V[KNB * KV_STRIDE];                   // 16896 B
    __shared__ float sh_hv[NH];
    __shared__ float sh_Q[NH];
    __shared__ float sh_att[NHEADS * KNB];
    __shared__ float sh_upd[NH];
    __shared__ float sh_mask[KNB];

    const int node = blockIdx.x;
    const int tid  = threadIdx.x;

    // ---- stage h_E[node] (f32 global, coalesced b128) -> LDS bf16 ----------
    const float* hEn = hE + (size_t)node * (KNB * NIN);
    for (int i = tid; i < (KNB * NIN) / 4; i += 256) {
        float4 v = ((const float4*)hEn)[i];
        int base = i * 4;
        int r = base / NIN;
        int c = base - r * NIN;                 // NIN % 4 == 0: same row
        __bf16* p = &sh_hE[r * HE_STRIDE + c];
        p[0] = (__bf16)v.x; p[1] = (__bf16)v.y;
        p[2] = (__bf16)v.z; p[3] = (__bf16)v.w;
    }
    if (tid < NH)  sh_hv[tid]   = hV[(size_t)node * NH + tid];
    if (tid < KNB) sh_mask[tid] = (float)mask[(size_t)node * KNB + tid];
    __syncthreads();

    // ---- Q = h_V row @ W_Q^T (128 dots of length 128, L2-hot weights) ------
    if (tid < NH) {
        float q = 0.f;
        #pragma unroll 8
        for (int j = 0; j < NH; ++j) q += sh_hv[j] * wqT[j * NH + tid];
        sh_Q[tid] = q;
    }

    // ---- K/V projections: [32 x 384] @ [384 x 128] via WMMA bf16 -----------
    // wave w owns output columns [16w, 16w+16); 2 M-tiles x 12 K-steps.
    const int w     = tid >> 5;
    const int lane  = tid & 31;
    const int khalf = lane >> 4;    // lane half selects K sub-range
    const int nrow  = lane & 15;    // A: row / B: column / D: column

    #pragma unroll
    for (int proj = 0; proj < 2; ++proj) {
        const __bf16* W = proj ? wvb : wkb;
        float* shOut    = proj ? sh_V : sh_K;
        v8f acc0 = {};                          // M-tile 0 (neighbors 0-15)
        v8f acc1 = {};                          // M-tile 1 (neighbors 16-31)
        #pragma unroll
        for (int ks = 0; ks < NIN / 32; ++ks) {
            // B fragment: W[n][k], 16 contiguous bf16 per lane (32B load)
            const __bf16* bp = &W[(w * 16 + nrow) * NIN + ks * 32 + khalf * 16];
            v16bf b = *(const v16bf*)bp;

            // A fragment, contiguous layout: elems 0-15 = K(16*khalf+0..15)
            //                                (matches B fragment order)
            const __bf16* ap0 = &sh_hE[nrow * HE_STRIDE + ks * 32 + 16 * khalf];
            v8bf a0lo = *(const v8bf*)ap0;
            v8bf a0hi = *(const v8bf*)(ap0 + 8);
            v16bf a0;
            #pragma unroll
            for (int e = 0; e < 8; ++e) { a0[e] = a0lo[e]; a0[e + 8] = a0hi[e]; }
            acc0 = __builtin_amdgcn_wmma_f32_16x16x32_bf16(
                       false, a0, false, b, (short)0, acc0, false, false);

            const __bf16* ap1 = ap0 + 16 * HE_STRIDE;
            v8bf a1lo = *(const v8bf*)ap1;
            v8bf a1hi = *(const v8bf*)(ap1 + 8);
            v16bf a1;
            #pragma unroll
            for (int e = 0; e < 8; ++e) { a1[e] = a1lo[e]; a1[e + 8] = a1hi[e]; }
            acc1 = __builtin_amdgcn_wmma_f32_16x16x32_bf16(
                       false, a1, false, b, (short)0, acc1, false, false);
        }
        // D layout: lane col = nrow, row = r + 8*khalf (+16 for tile 1)
        #pragma unroll
        for (int r = 0; r < 8; ++r) {
            shOut[(r + 8 * khalf) * KV_STRIDE + w * 16 + nrow]        = acc0[r];
            shOut[(16 + r + 8 * khalf) * KV_STRIDE + w * 16 + nrow]   = acc1[r];
        }
    }
    __syncthreads();

    // ---- masked softmax attention: wave h = head h, lane = neighbor k ------
    if (tid < NHEADS * KNB) {
        const int h = tid >> 5;
        const int k = tid & 31;
        const float* kr = &sh_K[k * KV_STRIDE + h * DH];
        const float* qr = &sh_Q[h * DH];
        float logit = 0.f;
        #pragma unroll
        for (int d = 0; d < DH; ++d) logit += qr[d] * kr[d];
        logit *= 0.17677669529663687f;          // 1/sqrt(32)
        const float msk = sh_mask[k];
        logit = (msk > 0.f) ? logit : -3.40282347e38f;

        float m = logit;                        // wave32 max-reduce
        #pragma unroll
        for (int off = 16; off > 0; off >>= 1) m = fmaxf(m, __shfl_xor(m, off));
        float e = __expf(logit - m);
        float s = e;                            // wave32 sum-reduce
        #pragma unroll
        for (int off = 16; off > 0; off >>= 1) s += __shfl_xor(s, off);
        sh_att[tid] = (e / s) * msk;
    }
    __syncthreads();

    // ---- h_upd[h,d] = sum_k att[h,k] * V[k, h*32+d] -------------------------
    if (tid < NH) {
        const int h = tid >> 5;
        const int d = tid & 31;
        float acc = 0.f;
        #pragma unroll 8
        for (int k = 0; k < KNB; ++k)
            acc += sh_att[h * KNB + k] * sh_V[k * KV_STRIDE + h * DH + d];
        sh_upd[tid] = acc;
    }
    __syncthreads();

    // ---- out = h_upd @ W_O^T (coalesced woT reads, broadcast sh_upd) -------
    if (tid < NH) {
        float o = 0.f;
        #pragma unroll 8
        for (int j = 0; j < NH; ++j) o += sh_upd[j] * woT[j * NH + tid];
        out[(size_t)node * NH + tid] = o;
    }
}

// ---------------------------------------------------------------------------
extern "C" void kernel_launch(void* const* d_in, const int* in_sizes, int n_in,
                              void* d_out, int out_size, void* d_ws, size_t ws_size,
                              hipStream_t stream) {
    const float* hV   = (const float*)d_in[0];
    const float* hE   = (const float*)d_in[1];
    const int*   mask = (const int*)d_in[2];
    const float* Wq   = (const float*)d_in[3];
    const float* Wk   = (const float*)d_in[4];
    const float* Wv   = (const float*)d_in[5];
    const float* Wo   = (const float*)d_in[6];
    float* out        = (float*)d_out;

    // workspace partition (total 320 KB)
    __bf16* wkb = (__bf16*)d_ws;                              //  98304 B
    __bf16* wvb = (__bf16*)((char*)d_ws + 98304);             //  98304 B
    float*  wqT = (float*)((char*)d_ws + 196608);             //  65536 B
    float*  woT = (float*)((char*)d_ws + 262144);             //  65536 B

    const int nodes = in_sizes[0] / NH;                       // B*N = 40000

    prep_weights<<<(NH * NIN + 255) / 256, 256, 0, stream>>>(
        Wq, Wk, Wv, Wo, wkb, wvb, wqT, woT);
    neigh_attn<<<nodes, 256, 0, stream>>>(
        hV, hE, mask, wkb, wvb, wqT, woT, out);
}
